// Tacotron2_59133109732073
// MI455X (gfx1250) — matrix operand order repaired and checked by reading, not compiled, not run on hardware
//
#include <hip/hip_runtime.h>
#include <math.h>

// ---------------- types ----------------
typedef __attribute__((ext_vector_type(16))) __bf16 v16bf;
typedef __attribute__((ext_vector_type(8)))  __bf16 v8bf;
typedef __attribute__((ext_vector_type(8)))  float  v8f;

#define B_   32
#define T_   768
#define NMEL 80
#define DPOS 544
#define PRE_ 256
#define RNN_ 1024
#define GATE 4096

__device__ __forceinline__ float sigm(float x) { return 1.0f / (1.0f + __expf(-x)); }

__device__ __forceinline__ v16bf cat16(v8bf lo, v8bf hi) {
    return __builtin_shufflevector(lo, hi, 0,1,2,3,4,5,6,7,8,9,10,11,12,13,14,15);
}

// One region of the K loop, compile-time stride and tile count.
// A layout per ISA 16-bit 16x32 tile: lane m / lane-group selects K-half:
//   lanes 0-15 : halves 0..7 -> K=kloc+0..7,  halves 8..15 -> K=kloc+16..23
//   lanes 16-31: halves 0..7 -> K=kloc+8..15, halves 8..15 -> K=kloc+24..31
template <int STRIDE, int TILES>
__device__ __forceinline__ void gemm_region(const __bf16* __restrict__ base,
                                            int col, int grp,
                                            const __bf16*& wp,
                                            v8f& acc0, v8f& acc1) {
    const __bf16* pa0 = base + (size_t)col * STRIDE + grp * 8;
    const __bf16* pa1 = base + (size_t)(col + 16) * STRIDE + grp * 8;
#pragma unroll 4
    for (int kt = 0; kt < TILES; ++kt) {
        v8bf a0lo = *(const v8bf*)(pa0);
        v8bf a0hi = *(const v8bf*)(pa0 + 16);
        v8bf a1lo = *(const v8bf*)(pa1);
        v8bf a1hi = *(const v8bf*)(pa1 + 16);
        v8bf blo  = *(const v8bf*)(wp);
        v8bf bhi  = *(const v8bf*)(wp + 8);
        v16bf a0 = cat16(a0lo, a0hi);
        v16bf a1 = cat16(a1lo, a1hi);
        v16bf bt = cat16(blo, bhi);
        acc0 = __builtin_amdgcn_wmma_f32_16x16x32_bf16(false, a0, false, bt,
                                                       (short)0, acc0, false, false);
        acc1 = __builtin_amdgcn_wmma_f32_16x16x32_bf16(false, a1, false, bt,
                                                       (short)0, acc1, false, false);
        pa0 += 32;
        pa1 += 32;
        wp  += 512;
    }
}

// -------- per-step gate GEMM: Z(32 x 4096) = concat(A0|A1|A2)(32 x K) @ Wp + bias
template <int S0, int T0, int S1, int T1, int S2, int T2>
__global__ void gemm_wmma(const __bf16* __restrict__ A0,
                          const __bf16* __restrict__ A1r,
                          const __bf16* __restrict__ A2r,
                          const __bf16* __restrict__ Wp,
                          const float* __restrict__ bias,
                          float* __restrict__ Z) {
    int lane = threadIdx.x & 31;
    int wave = threadIdx.x >> 5;
    int nt = blockIdx.x * 4 + wave;
    constexpr int KT = T0 + T1 + T2;
    int grp = lane >> 4;
    int col = lane & 15;
    int n = nt * 16 + col;
    float bv = bias[n];
    v8f acc0 = {};
    v8f acc1 = {};
    const __bf16* wp = Wp + ((size_t)nt * KT * 32 + lane) * 16;
    gemm_region<S0, T0>(A0,  col, grp, wp, acc0, acc1);
    gemm_region<S1, T1>(A1r, col, grp, wp, acc0, acc1);
    if (T2 > 0) gemm_region<S2, T2>(A2r, col, grp, wp, acc0, acc1);
#pragma unroll
    for (int v = 0; v < 8; ++v) {
        int m0 = v + grp * 8;
        Z[(size_t)m0 * GATE + n]        = acc0[v] + bv;
        Z[(size_t)(m0 + 16) * GATE + n] = acc1[v] + bv;
    }
}

// -------- projection: out[:, :, t] = concat(h2, dur_t) @ proj_w + proj_b --------
__global__ void proj_wmma(const __bf16* __restrict__ h2,
                          const __bf16* __restrict__ durt,
                          const __bf16* __restrict__ Wp,
                          const float* __restrict__ bias,
                          float* __restrict__ out, int t) {
    int lane = threadIdx.x & 31;
    int nt = blockIdx.x;             // 0..4  (80 = 5 * 16)
    constexpr int KT = 49;           // 32 (h2) + 17 (dur)
    int grp = lane >> 4;
    int col = lane & 15;
    v8f acc0 = {};
    v8f acc1 = {};
    const __bf16* wp = Wp + ((size_t)nt * KT * 32 + lane) * 16;
    gemm_region<RNN_, 32>(h2,   col, grp, wp, acc0, acc1);
    gemm_region<DPOS, 17>(durt, col, grp, wp, acc0, acc1);
    int n = nt * 16 + col;
    float bv = bias[n];
#pragma unroll
    for (int v = 0; v < 8; ++v) {
        int m0 = v + grp * 8;
        out[((size_t)m0 * NMEL + n) * T_ + t]        = acc0[v] + bv;
        out[((size_t)(m0 + 16) * NMEL + n) * T_ + t] = acc1[v] + bv;
    }
}

// ---------------- weight packing (once per call, f32 -> bf16 WMMA-B layout) ---
// Packed layout: tile (nt,kt) -> 32 lanes * 16 halves contiguous.
// Lane L element s=2r+h supplies B[K = kt*32 + (L/16)*8 + (s<8 ? s : s+8)][N = nt*16 + (L%16)].
// Source is the logical row-major concat [A(rowsA) ; Bsrc] of width N.
__global__ void pack_w(const float* __restrict__ A, int rowsA,
                       const float* __restrict__ Bsrc,
                       int N, int KT, int NT, __bf16* __restrict__ dst) {
    int id = blockIdx.x * 256 + threadIdx.x;
    int total = KT * NT * 32;
    if (id >= total) return;
    int lane = id & 31;
    int tile = id >> 5;
    int kt = tile % KT;
    int nt = tile / KT;
    int n = nt * 16 + (lane & 15);
    __bf16* d = dst + (size_t)id * 16;
    for (int r = 0; r < 8; ++r)
        for (int h = 0; h < 2; ++h) {
            int k = kt * 32 + (lane >> 4) * 8 + ((2 * r + h) < 8 ? (2 * r + h) : (2 * r + h + 8));
            float v = (k < rowsA) ? A[(size_t)k * N + n]
                                  : Bsrc[(size_t)(k - rowsA) * N + n];
            d[r * 2 + h] = (__bf16)v;
        }
}

// dur (B,T,DPOS) f32 -> (T,B,DPOS) bf16
__global__ void dur_to_bf16(const float* __restrict__ src, __bf16* __restrict__ dst) {
    int id = blockIdx.x * 256 + threadIdx.x;
    if (id >= T_ * B_ * DPOS) return;
    int t = id / (B_ * DPOS);
    int r = id % (B_ * DPOS);
    int b = r / DPOS;
    int k = r % DPOS;
    dst[id] = (__bf16)src[((size_t)b * T_ + t) * DPOS + k];
}

// Prenet for all timesteps in parallel: pre[tau][b][:] ; tau==0 row is zeros input.
__global__ void prenet_kernel(const float* __restrict__ dec,
                              const float* __restrict__ w1,
                              const float* __restrict__ w2,
                              __bf16* __restrict__ pre) {
    __shared__ float xin[NMEL];
    __shared__ float x1[PRE_];
    int bid = blockIdx.x;            // tau*32 + b
    int tau = bid >> 5;
    int b   = bid & 31;
    int j   = threadIdx.x;           // 0..255
    if (j < NMEL)
        xin[j] = (tau == 0) ? 0.0f : dec[((size_t)b * NMEL + j) * T_ + (tau - 1)];
    __syncthreads();
    float a = 0.0f;
    for (int m = 0; m < NMEL; ++m) a += xin[m] * w1[m * PRE_ + j];
    x1[j] = fmaxf(a, 0.0f);
    __syncthreads();
    float a2 = 0.0f;
    for (int k = 0; k < PRE_; ++k) a2 += x1[k] * w2[k * PRE_ + j];
    pre[(size_t)bid * PRE_ + j] = (__bf16)fmaxf(a2, 0.0f);
}

__global__ void init_state(__bf16* h1, __bf16* h2, float* c1, float* c2) {
    int i = blockIdx.x * 256 + threadIdx.x;
    if (i < B_ * RNN_) {
        h1[i] = (__bf16)0.0f;
        h2[i] = (__bf16)0.0f;
        c1[i] = 0.0f;
        c2[i] = 0.0f;
    }
}

// -------- LayerNorm-LSTM cell update: one workgroup per batch row --------
__global__ void lstm_cell(const float* __restrict__ Z,
                          float* __restrict__ C,
                          __bf16* __restrict__ Hb,
                          const float* __restrict__ g,  const float* __restrict__ bn,
                          const float* __restrict__ gc, const float* __restrict__ bc) {
    __shared__ float red[256];
    __shared__ float red2[256];
    __shared__ float cbuf[RNN_];
    __shared__ float obuf[RNN_];
    int b = blockIdx.x;
    int tid = threadIdx.x;
    const float* z = Z + (size_t)b * GATE;

    // LN over the 4096-wide pre-activation
    float s = 0.0f, ss = 0.0f;
#pragma unroll
    for (int q = 0; q < 16; ++q) {
        float v = z[tid + q * 256];
        s += v; ss += v * v;
    }
    red[tid] = s; red2[tid] = ss;
    __syncthreads();
    for (int off = 128; off > 0; off >>= 1) {
        if (tid < off) { red[tid] += red[tid + off]; red2[tid] += red2[tid + off]; }
        __syncthreads();
    }
    float mean = red[0] * (1.0f / GATE);
    float var  = red2[0] * (1.0f / GATE) - mean * mean;
    float rstd = rsqrtf(var + 1e-5f);
    __syncthreads();

    // gates + c update; accumulate stats for the c-LayerNorm
    float s2 = 0.0f, ss2 = 0.0f;
#pragma unroll
    for (int q = 0; q < 4; ++q) {
        int j = tid + q * 256;
        float ni = (z[j]          - mean) * rstd * g[j]          + bn[j];
        float nf = (z[RNN_ + j]   - mean) * rstd * g[RNN_ + j]   + bn[RNN_ + j];
        float ng = (z[2*RNN_ + j] - mean) * rstd * g[2*RNN_ + j] + bn[2*RNN_ + j];
        float no = (z[3*RNN_ + j] - mean) * rstd * g[3*RNN_ + j] + bn[3*RNN_ + j];
        float cn = sigm(nf) * C[(size_t)b * RNN_ + j] + sigm(ni) * tanhf(ng);
        cbuf[j] = cn;
        obuf[j] = no;
        s2 += cn; ss2 += cn * cn;
    }
    __syncthreads();
    red[tid] = s2; red2[tid] = ss2;
    __syncthreads();
    for (int off = 128; off > 0; off >>= 1) {
        if (tid < off) { red[tid] += red[tid + off]; red2[tid] += red2[tid + off]; }
        __syncthreads();
    }
    float mean2 = red[0] * (1.0f / RNN_);
    float var2  = red2[0] * (1.0f / RNN_) - mean2 * mean2;
    float rstd2 = rsqrtf(var2 + 1e-5f);
    __syncthreads();

#pragma unroll
    for (int q = 0; q < 4; ++q) {
        int j = tid + q * 256;
        float cn = cbuf[j];
        C[(size_t)b * RNN_ + j] = cn;
        float hh = sigm(obuf[j]) * tanhf((cn - mean2) * rstd2 * gc[j] + bc[j]);
        Hb[(size_t)b * RNN_ + j] = (__bf16)hh;
    }
}

// ---------------- host side ----------------
extern "C" void kernel_launch(void* const* d_in, const int* in_sizes, int n_in,
                              void* d_out, int out_size, void* d_ws, size_t ws_size,
                              hipStream_t stream) {
    (void)in_sizes; (void)n_in; (void)out_size; (void)ws_size;
    const float* dur_in = (const float*)d_in[0];
    const float* dec_in = (const float*)d_in[1];
    // d_in[2] = memory_lengths (unused by reference)
    const float* pw1 = (const float*)d_in[3];
    const float* pw2 = (const float*)d_in[4];
    const float* w1x = (const float*)d_in[5];
    const float* w1h = (const float*)d_in[6];
    const float* b1  = (const float*)d_in[7];
    const float* g1  = (const float*)d_in[8];
    const float* bn1 = (const float*)d_in[9];
    const float* gc1 = (const float*)d_in[10];
    const float* bc1 = (const float*)d_in[11];
    const float* w2x = (const float*)d_in[12];
    const float* w2h = (const float*)d_in[13];
    const float* b2  = (const float*)d_in[14];
    const float* g2  = (const float*)d_in[15];
    const float* bn2 = (const float*)d_in[16];
    const float* gc2 = (const float*)d_in[17];
    const float* bc2 = (const float*)d_in[18];
    const float* pjw = (const float*)d_in[19];
    const float* pjb = (const float*)d_in[20];
    float* out = (float*)d_out;

    // workspace carve-up
    char* ws = (char*)d_ws;
    auto carve = [&](size_t bytes) {
        char* p = ws;
        ws += (bytes + 255) & ~(size_t)255;
        return p;
    };
    const int KT1 = 57;   // 8 (pre) + 17 (dur) + 32 (h1) tiles of K=32 -> 1824
    const int KT2 = 64;   // 32 (h1) + 32 (h2)                         -> 2048
    const int KTP = 49;   // 32 (h2) + 17 (dur)                        -> 1568
    const int NT  = GATE / 16;  // 256
    const int NTP = NMEL / 16;  // 5
    __bf16* W1p  = (__bf16*)carve((size_t)KT1 * NT  * 512 * 2);
    __bf16* W2p  = (__bf16*)carve((size_t)KT2 * NT  * 512 * 2);
    __bf16* Wpp  = (__bf16*)carve((size_t)KTP * NTP * 512 * 2);
    __bf16* preb = (__bf16*)carve((size_t)T_ * B_ * PRE_ * 2);
    __bf16* durb = (__bf16*)carve((size_t)T_ * B_ * DPOS * 2);
    __bf16* h1b  = (__bf16*)carve((size_t)B_ * RNN_ * 2);
    __bf16* h2b  = (__bf16*)carve((size_t)B_ * RNN_ * 2);
    float*  c1   = (float*)carve((size_t)B_ * RNN_ * 4);
    float*  c2   = (float*)carve((size_t)B_ * RNN_ * 4);
    float*  z1   = (float*)carve((size_t)B_ * GATE * 4);
    float*  z2   = (float*)carve((size_t)B_ * GATE * 4);

    // one-time (per call) prep, all parallel over T
    init_state<<<(B_ * RNN_ + 255) / 256, 256, 0, stream>>>(h1b, h2b, c1, c2);
    {
        int tot = KT1 * NT * 32;
        pack_w<<<(tot + 255) / 256, 256, 0, stream>>>(w1x, 800, w1h, GATE, KT1, NT, W1p);
    }
    {
        int tot = KT2 * NT * 32;
        pack_w<<<(tot + 255) / 256, 256, 0, stream>>>(w2x, 1024, w2h, GATE, KT2, NT, W2p);
    }
    {
        int tot = KTP * NTP * 32;
        pack_w<<<(tot + 255) / 256, 256, 0, stream>>>(pjw, 1568, pjw, NMEL, KTP, NTP, Wpp);
    }
    {
        int tot = T_ * B_ * DPOS;
        dur_to_bf16<<<(tot + 255) / 256, 256, 0, stream>>>(dur_in, durb);
    }
    prenet_kernel<<<T_ * B_, 256, 0, stream>>>(dec_in, pw1, pw2, preb);

    // sequential scan: 5 launches per step
    for (int t = 0; t < T_; ++t) {
        const __bf16* pre_t = preb + (size_t)t * B_ * PRE_;
        const __bf16* dur_t = durb + (size_t)t * B_ * DPOS;
        gemm_wmma<PRE_, 8, DPOS, 17, RNN_, 32><<<NT / 4, 128, 0, stream>>>(
            pre_t, dur_t, h1b, W1p, b1, z1);
        lstm_cell<<<B_, 256, 0, stream>>>(z1, c1, h1b, g1, bn1, gc1, bc1);
        gemm_wmma<RNN_, 32, RNN_, 32, RNN_, 0><<<NT / 4, 128, 0, stream>>>(
            h1b, h2b, h2b, W2p, b2, z2);
        lstm_cell<<<B_, 256, 0, stream>>>(z2, c2, h2b, g2, bn2, gc2, bc2);
        proj_wmma<<<NTP, 32, 0, stream>>>(h2b, dur_t, Wpp, pjb, out, t);
    }
}
